// PHGNS2_19748259627134
// MI455X (gfx1250) — matrix operand order repaired and checked by reading, not compiled, not run on hardware
//
// PHGNS2 port-Hamiltonian GNN integrator for MI455X (gfx1250, wave32, WMMA).
// GEMMs: v_wmma_f32_16x16x32_bf16 (bf16 in / f32 acc) with fused bias/ReLU/LayerNorm/
// residual epilogues. Weights pre-converted to bf16 (n-major) once per launch so the
// B tile stages via global_load_async_to_lds_b128 (ASYNCcnt).
// Matvecs (P_inv/L_inv/U_inv, ~0.74GB streamed) are the HBM-bound part (~32us @23.3TB/s).
#include <hip/hip_runtime.h>
#include <hip/hip_bf16.h>

#define NQv     2048
#define NNODEv  2048
#define NRv     2048
#define LATv    128
#define NDIFFv  4096
#define NSTATEv 6144
#define DTv     0.01f

typedef __attribute__((ext_vector_type(16))) __bf16         bf16x16;
typedef __attribute__((ext_vector_type(8)))  __bf16         bf16x8;
typedef __attribute__((ext_vector_type(8)))  float          f32x8;
typedef __attribute__((ext_vector_type(8)))  unsigned short us8;
typedef __attribute__((ext_vector_type(4)))  int            i32x4;

#define AS1 __attribute__((address_space(1)))
#define AS3 __attribute__((address_space(3)))

#if __has_builtin(__builtin_amdgcn_global_load_async_to_lds_b128) && \
    __has_builtin(__builtin_amdgcn_s_wait_asynccnt)
#define HAVE_ASYNC_LDS 1
#else
#define HAVE_ASYNC_LDS 0
#endif

// ---------------------------------------------------------------- utilities
__device__ inline unsigned short f2bf(float f) {   // fp32 -> bf16 (RNE)
    unsigned u = __float_as_uint(f);
    u += 0x7fffu + ((u >> 16) & 1u);
    return (unsigned short)(u >> 16);
}

__device__ inline float block_sum(float v) {
    __shared__ float red[9];
    #pragma unroll
    for (int off = 16; off > 0; off >>= 1) v += __shfl_xor(v, off, 32);
    const int wv = threadIdx.x >> 5, ln = threadIdx.x & 31;
    const int nw = blockDim.x >> 5;
    __syncthreads();
    if (ln == 0) red[wv] = v;
    __syncthreads();
    if (threadIdx.x == 0) {
        float s = 0.f;
        for (int i = 0; i < nw; ++i) s += red[i];
        red[8] = s;
    }
    __syncthreads();
    return red[8];
}

// weight prep: bf16 transpose (forward: W[KxN] -> Wt[n*K+k]) and bf16 copy (backward)
__global__ void k_w2bf_t(const float* __restrict__ W, unsigned short* __restrict__ Wt,
                         int K, int N) {
    int i = blockIdx.x * 256 + threadIdx.x;
    if (i >= K * N) return;
    int n = i / K, k = i - n * K;
    Wt[i] = f2bf(W[(size_t)k * N + n]);
}
__global__ void k_w2bf(const float* __restrict__ W, unsigned short* __restrict__ Wo, int n) {
    int i = blockIdx.x * 256 + threadIdx.x;
    if (i < n) Wo[i] = f2bf(W[i]);
}

// ---------------------------------------------------------------- WMMA GEMM
// C[M x N] = A[M x K] * B, where B is supplied n-major bf16: Bt[n*ldbt + k] = B[k][n].
// Block: 256 thr (8 waves), tile 128x128, wave = 16 rows x 128 cols (8 D-tiles).
// Epilogue: +bias, ReLU, relu-mask (grad), LayerNorm (full row in-wave, stash y/rstd),
// residual add.  Runtime-nullable pointers; LN only used when N==128.
template <bool RELU, bool LN>
__global__ __launch_bounds__(256) void k_gemm(
    const float* __restrict__ A, int lda,
    const unsigned short* __restrict__ Bt, int ldbt,
    const float* __restrict__ bias,
    const float* __restrict__ res,
    const float* __restrict__ mask,
    float* __restrict__ out, int ldout,
    float* __restrict__ stash_y, float* __restrict__ stash_rstd,
    int K)
{
    __shared__ unsigned short sA[128 * 40];   // [m][k] bf16, stride 40 => b128-friendly
    __shared__ unsigned short sB[128 * 40];   // [n][k] bf16

    const int tid  = threadIdx.x;
    const int wave = tid >> 5;
    const int lane = tid & 31;
    const int m0   = blockIdx.x * 128;
    const int n0   = blockIdx.y * 128;

    f32x8 acc[8];
    #pragma unroll
    for (int t = 0; t < 8; ++t)
        #pragma unroll
        for (int r = 0; r < 8; ++r) acc[t][r] = 0.f;

    for (int k0 = 0; k0 < K; k0 += 32) {
        // B tile: 128 rows x 32 bf16 = contiguous 16B segments -> async DMA to LDS
#if HAVE_ASYNC_LDS
        #pragma unroll
        for (int i = 0; i < 2; ++i) {
            int idx = tid + i * 256;                 // 512 x 16B segments
            int row = idx >> 2, seg = (idx & 3) << 3;
            __builtin_amdgcn_global_load_async_to_lds_b128(
                (AS1 i32x4*)(Bt + (size_t)(n0 + row) * ldbt + k0 + seg),
                (AS3 i32x4*)&sB[row * 40 + seg], 0, 0);
        }
#else
        #pragma unroll
        for (int i = 0; i < 2; ++i) {
            int idx = tid + i * 256;
            int row = idx >> 2, seg = (idx & 3) << 3;
            *(us8*)&sB[row * 40 + seg] =
                *(const us8*)(Bt + (size_t)(n0 + row) * ldbt + k0 + seg);
        }
#endif
        // A tile: 128x32 fp32 -> bf16, packed 8-wide => ds_store_b128
        #pragma unroll
        for (int i = 0; i < 2; ++i) {
            int idx = tid + i * 256;                 // 512 x 8-elem segments
            int r = idx >> 2, c8 = (idx & 3) << 3;
            const float* src = A + (size_t)(m0 + r) * lda + k0 + c8;
            float4 v0 = *(const float4*)src;
            float4 v1 = *(const float4*)(src + 4);
            us8 pk = { f2bf(v0.x), f2bf(v0.y), f2bf(v0.z), f2bf(v0.w),
                       f2bf(v1.x), f2bf(v1.y), f2bf(v1.z), f2bf(v1.w) };
            *(us8*)&sA[r * 40 + c8] = pk;
        }
#if HAVE_ASYNC_LDS
        __builtin_amdgcn_s_wait_asynccnt(0);
#endif
        __syncthreads();

        // Pack A operand (16x32 bf16 lane layout: lanes<16 K0..7|K16..23, >=16 +8)
        const int mloc = wave * 16 + (lane & 15);
        const int kb   = (lane >> 4) * 8;
        bf16x8 alo = *(const bf16x8*)&sA[mloc * 40 + kb];
        bf16x8 ahi = *(const bf16x8*)&sA[mloc * 40 + 16 + kb];
        bf16x16 aop = __builtin_shufflevector(alo, ahi,
            0,1,2,3,4,5,6,7,8,9,10,11,12,13,14,15);

        const int kb2 = (lane >> 4) * 8;    // B 32x16: lanes<16 K0..7|K16..23, >=16 +8
        #pragma unroll
        for (int t = 0; t < 8; ++t) {
            const int nloc = t * 16 + (lane & 15);
            bf16x8 blo = *(const bf16x8*)&sB[nloc * 40 + kb2];
            bf16x8 bhi = *(const bf16x8*)&sB[nloc * 40 + kb2 + 16];
            bf16x16 bop = __builtin_shufflevector(blo, bhi,
                0,1,2,3,4,5,6,7,8,9,10,11,12,13,14,15);
            acc[t] = __builtin_amdgcn_wmma_f32_16x16x32_bf16(
                false, aop, false, bop, (short)0, acc[t], false, false);
        }
        __syncthreads();
    }

    // --------------------------------------------------------- epilogue
    const int half  = lane >> 4;
    const int c0    = lane & 15;
    const int mrow0 = m0 + wave * 16 + half * 8;

    if (bias) {
        #pragma unroll
        for (int t = 0; t < 8; ++t) {
            const float bv = bias[n0 + t * 16 + c0];
            #pragma unroll
            for (int r = 0; r < 8; ++r) acc[t][r] += bv;
        }
    }
    if (RELU) {
        #pragma unroll
        for (int t = 0; t < 8; ++t)
            #pragma unroll
            for (int r = 0; r < 8; ++r) acc[t][r] = fmaxf(acc[t][r], 0.f);
    }
    if (mask) {   // grad through ReLU: zero where stored activation <= 0
        #pragma unroll
        for (int t = 0; t < 8; ++t)
            #pragma unroll
            for (int r = 0; r < 8; ++r) {
                size_t o = (size_t)(mrow0 + r) * ldout + n0 + t * 16 + c0;
                if (!(mask[o] > 0.f)) acc[t][r] = 0.f;
            }
    }
    if (LN) {     // full 128-wide row lives in this wave's half: shuffle-reduce
        #pragma unroll
        for (int r = 0; r < 8; ++r) {
            float s1 = 0.f, s2 = 0.f;
            #pragma unroll
            for (int t = 0; t < 8; ++t) { float x = acc[t][r]; s1 += x; s2 += x * x; }
            #pragma unroll
            for (int off = 1; off < 16; off <<= 1) {
                s1 += __shfl_xor(s1, off, 32);
                s2 += __shfl_xor(s2, off, 32);
            }
            const float mu = s1 * 0.0078125f;
            const float rs = rsqrtf(s2 * 0.0078125f - mu * mu + 1e-6f);
            #pragma unroll
            for (int t = 0; t < 8; ++t) acc[t][r] = (acc[t][r] - mu) * rs;
            if (stash_rstd && c0 == 0) stash_rstd[mrow0 + r] = rs;
        }
    }
    #pragma unroll
    for (int t = 0; t < 8; ++t)
        #pragma unroll
        for (int r = 0; r < 8; ++r) {
            const size_t o = (size_t)(mrow0 + r) * ldout + n0 + t * 16 + c0;
            float yv = acc[t][r];
            if (LN && stash_y) stash_y[o] = yv;
            if (res) yv += res[o];
            out[o] = yv;
        }
}

// ---------------------------------------------------------------- small kernels
__global__ void k_affine1(const float* __restrict__ ef, const float* __restrict__ w,
                          const float* __restrict__ b, float* __restrict__ out, int M) {
    int i = blockIdx.x * 256 + threadIdx.x;
    if (i >= M * LATv) return;
    int m = i >> 7, n = i & 127;
    out[i] = fmaxf(ef[m] * w[n] + b[n], 0.f);
}

__global__ void k_seed_dec(const float* __restrict__ h2, const float* __restrict__ w2,
                           float* __restrict__ g, int M) {
    int i = blockIdx.x * 256 + threadIdx.x;
    if (i >= M * LATv) return;
    g[i] = (h2[i] > 0.f) ? w2[i & 127] : 0.f;
}

__global__ __launch_bounds__(128) void k_rowdot(const float* __restrict__ A,
                                                const float* __restrict__ w,
                                                const float* __restrict__ b,
                                                float* __restrict__ out) {
    int row = blockIdx.x;
    float v = A[(size_t)row * 128 + threadIdx.x] * w[threadIdx.x];
    float s = block_sum(v);
    if (threadIdx.x == 0) out[row] = s + (b ? b[0] : 0.f);
}

__global__ __launch_bounds__(128) void k_ln_bwd(const float* __restrict__ y,
                                                const float* __restrict__ rstd,
                                                const float* __restrict__ g,
                                                float* __restrict__ out) {
    int row = blockIdx.x, j = threadIdx.x;
    float gv = g[(size_t)row * 128 + j], yv = y[(size_t)row * 128 + j];
    float s1 = block_sum(gv);
    float s2 = block_sum(gv * yv);
    out[(size_t)row * 128 + j] =
        rstd[row] * (gv - s1 * 0.0078125f - yv * (s2 * 0.0078125f));
}

__global__ __launch_bounds__(256) void k_matvec(const float* __restrict__ A,
                                                const float* __restrict__ x,
                                                const float* __restrict__ base,
                                                float alpha, int N, float* __restrict__ y) {
    const float* ar = A + (size_t)blockIdx.x * N;
    float acc = 0.f;
    for (int j = threadIdx.x * 4; j < N; j += 1024) {
        __builtin_prefetch(ar + j + 4096, 0, 1);
        float4 a = *(const float4*)(ar + j);
        float4 xv = *(const float4*)(x + j);
        acc += a.x * xv.x + a.y * xv.y + a.z * xv.z + a.w * xv.w;
    }
    float s = block_sum(acc);
    if (threadIdx.x == 0) y[blockIdx.x] = alpha * s + (base ? base[blockIdx.x] : 0.f);
}

__global__ __launch_bounds__(128) void k_matvec_t(const float* __restrict__ A,
                                                  const float* __restrict__ x,
                                                  float alpha, int M, int N,
                                                  float* __restrict__ y) {
    __shared__ float sx[128];
    int col = blockIdx.x * 128 + threadIdx.x;
    float acc = 0.f;
    for (int m0 = 0; m0 < M; m0 += 128) {
        __syncthreads();
        sx[threadIdx.x] = x[m0 + threadIdx.x];
        __syncthreads();
        #pragma unroll 4
        for (int m = 0; m < 128; ++m) acc += A[(size_t)(m0 + m) * N + col] * sx[m];
    }
    y[col] = alpha * acc;
}

__global__ void k_build_msg(const float* __restrict__ e, const float* __restrict__ nl,
                            const int* __restrict__ s, const int* __restrict__ r,
                            float* __restrict__ msg) {
    int i = blockIdx.x * 256 + threadIdx.x;
    if (i >= NDIFFv * LATv) return;
    int m = i >> 7, c = i & 127;
    msg[(size_t)m * 384 + c]       = e[i];
    msg[(size_t)m * 384 + 128 + c] = nl[(size_t)s[m] * 128 + c];
    msg[(size_t)m * 384 + 256 + c] = nl[(size_t)r[m] * 128 + c];
}

__global__ void k_build_cat(const float* __restrict__ nl, const float* __restrict__ agg,
                            float* __restrict__ cat) {
    int i = blockIdx.x * 256 + threadIdx.x;
    if (i >= NNODEv * LATv) return;
    int m = i >> 7, c = i & 127;
    cat[(size_t)m * 256 + c]       = nl[i];
    cat[(size_t)m * 256 + 128 + c] = agg[i];
}

// dst[idx[m]*128+c] += src[m*ld+off+c]  (fp32 atomics; order nondeterministic bitwise)
__global__ void k_scatter_add(const float* __restrict__ src, int ld, int off,
                              const int* __restrict__ idx, float* __restrict__ dst, int M) {
    int i = blockIdx.x * 256 + threadIdx.x;
    if (i >= M * LATv) return;
    int m = i >> 7, c = i & 127;
    atomicAdd(&dst[(size_t)idx[m] * 128 + c], src[(size_t)m * ld + off + c]);
}

__global__ void k_gather_add(float* __restrict__ dst, const float* __restrict__ src,
                             int ld, int off, const int* __restrict__ idx, int M) {
    int i = blockIdx.x * 256 + threadIdx.x;
    if (i >= M * LATv) return;
    int m = i >> 7, c = i & 127;
    dst[i] += src[(size_t)idx[m] * ld + off + c];
}

__global__ void k_add_slice(float* __restrict__ dst, const float* __restrict__ src,
                            int ld, int off, int M) {
    int i = blockIdx.x * 256 + threadIdx.x;
    if (i >= M * LATv) return;
    int m = i >> 7, c = i & 127;
    dst[i] += src[(size_t)m * ld + off + c];
}

__global__ void k_copy(float* __restrict__ d, const float* __restrict__ s, int n) {
    int i = blockIdx.x * 256 + threadIdx.x;
    if (i < n) d[i] = s[i];
}
__global__ void k_sub(float* __restrict__ d, const float* __restrict__ a,
                      const float* __restrict__ b, int n) {
    int i = blockIdx.x * 256 + threadIdx.x;
    if (i < n) d[i] = a[i] - b[i];
}
__global__ void k_abs_sub(float* __restrict__ d, const float* __restrict__ a,
                          const float* __restrict__ b, int n) {
    int i = blockIdx.x * 256 + threadIdx.x;
    if (i < n) d[i] = fabsf(a[i] - b[i]);
}
__global__ __launch_bounds__(256) void k_sum(const float* __restrict__ x, int n,
                                             float* __restrict__ out) {
    float a = 0.f;
    for (int i = threadIdx.x; i < n; i += 256) a += x[i];
    float s = block_sum(a);
    if (threadIdx.x == 0) out[0] = s;
}

// ---------------------------------------------------------------- host orchestration
struct MLP2 { const float *W0, *b0, *W1, *b1; };
struct MLP3 { const float *W0, *b0, *W1, *b1, *W2, *b2; };

extern "C" void kernel_launch(void* const* d_in, const int* in_sizes, int n_in,
                              void* d_out, int out_size, void* d_ws, size_t ws_size,
                              hipStream_t stream) {
    (void)in_sizes; (void)n_in; (void)out_size; (void)ws_size;
    int p = 0;
    const float* edges = (const float*)d_in[p++];   // 4096  (ef initial)
    const float* nodes = (const float*)d_in[p++];   // 2048
    p++;                                            // control (unused)
    const int* senders   = (const int*)d_in[p++];
    const int* receivers = (const int*)d_in[p++];
    auto m2 = [&]() { MLP2 m; m.W0=(const float*)d_in[p++]; m.b0=(const float*)d_in[p++];
                      m.W1=(const float*)d_in[p++]; m.b1=(const float*)d_in[p++]; return m; };
    auto m3 = [&]() { MLP3 m; m.W0=(const float*)d_in[p++]; m.b0=(const float*)d_in[p++];
                      m.W1=(const float*)d_in[p++]; m.b1=(const float*)d_in[p++];
                      m.W2=(const float*)d_in[p++]; m.b2=(const float*)d_in[p++]; return m; };
    MLP2 enc_e = m2(), enc_n = m2(), mp_e = m2(), mp_n = m2();
    MLP3 dec_e = m3(), dec_n = m3();
    const float* AC = (const float*)d_in[p++];
    const float* AR = (const float*)d_in[p++];
    const float* AL = (const float*)d_in[p++];
    p++;                                            // AI (zeros; -AI@i_src == 0)
    const float* P_inv = (const float*)d_in[p++];
    const float* L_inv = (const float*)d_in[p++];
    const float* U_inv = (const float*)d_in[p++];
    float* out = (float*)d_out;

    // workspace bump allocator (~70 MB)
    char* wp = (char*)d_ws;
    auto alloc_b = [&](size_t bytes) { void* r = (void*)wp;
                                       wp += ((bytes + 255) / 256) * 256; return r; };
    auto alloc   = [&](size_t n) { return (float*)alloc_b(n * 4); };
    auto alloc_u = [&](size_t n) { return (unsigned short*)alloc_b(n * 2); };

    float* h1_enc = alloc(NDIFFv*LATv); float* y_enc = alloc(NDIFFv*LATv);
    float* rstd_enc = alloc(NDIFFv);
    float *h1_e[5], *y_e[5], *rstd_e[5], *h1_n[5], *y_n[5], *rstd_n[5];
    for (int i = 0; i < 5; ++i) {
        h1_e[i]=alloc(NDIFFv*LATv); y_e[i]=alloc(NDIFFv*LATv); rstd_e[i]=alloc(NDIFFv);
        h1_n[i]=alloc(NNODEv*LATv); y_n[i]=alloc(NNODEv*LATv); rstd_n[i]=alloc(NNODEv);
    }
    float* h1_d = alloc(NDIFFv*LATv); float* h2_d = alloc(NDIFFv*LATv);
    float* eA = alloc(NDIFFv*LATv);  float* eB = alloc(NDIFFv*LATv);
    float* nA = alloc(NNODEv*LATv);  float* nB = alloc(NNODEv*LATv);
    float* msg = alloc(NDIFFv*384);  float* cat = alloc(NNODEv*256);
    float* agg = alloc(NNODEv*LATv);
    float* tmp1 = alloc(NDIFFv*LATv); float* tmp2 = alloc(NDIFFv*LATv);
    float* g_e = alloc(NDIFFv*LATv);  float* g_n = alloc(NNODEv*LATv);
    float* dec_ev = alloc(NDIFFv);    float* next_y = alloc(NNODEv);
    float* dH  = alloc(NDIFFv);
    float* fvec = alloc(NSTATEv); float* hvec = alloc(NSTATEv); float* wvec = alloc(NSTATEv);
    float* x1 = alloc(NDIFFv); float* x2 = alloc(NDIFFv);
    float* t_r = alloc(NRv); float* F0v = alloc(NNODEv);
    float* F1v = alloc(NNODEv); float* ACtv = alloc(NQv);

    auto ew = [](int n) { return dim3((n + 255) / 256); };

    // bf16 weight prep (once per launch): Wt = transpose for forward, Wo = copy for bwd
    auto prep_t = [&](const float* W, int K, int N) {
        unsigned short* d = alloc_u((size_t)K * N);
        k_w2bf_t<<<ew(K*N),256,0,stream>>>(W, d, K, N);
        return d;
    };
    auto prep_c = [&](const float* W, int n) {
        unsigned short* d = alloc_u(n);
        k_w2bf<<<ew(n),256,0,stream>>>(W, d, n);
        return d;
    };
    unsigned short* enc_eW1t = prep_t(enc_e.W1, 128, 128);
    unsigned short* enc_nW1t = prep_t(enc_n.W1, 128, 128);
    unsigned short* mp_eW0t  = prep_t(mp_e.W0, 384, 128);
    unsigned short* mp_eW1t  = prep_t(mp_e.W1, 128, 128);
    unsigned short* mp_nW0t  = prep_t(mp_n.W0, 256, 128);
    unsigned short* mp_nW1t  = prep_t(mp_n.W1, 128, 128);
    unsigned short* dec_eW0t = prep_t(dec_e.W0, 128, 128);
    unsigned short* dec_eW1t = prep_t(dec_e.W1, 128, 128);
    unsigned short* dec_nW0t = prep_t(dec_n.W0, 128, 128);
    unsigned short* dec_nW1t = prep_t(dec_n.W1, 128, 128);
    unsigned short* enc_eW1o = prep_c(enc_e.W1, 128*128);
    unsigned short* mp_eW0o  = prep_c(mp_e.W0, 384*128);
    unsigned short* mp_eW1o  = prep_c(mp_e.W1, 128*128);
    unsigned short* mp_nW0o  = prep_c(mp_n.W0, 256*128);
    unsigned short* mp_nW1o  = prep_c(mp_n.W1, 128*128);
    unsigned short* dec_eW0o = prep_c(dec_e.W0, 128*128);
    unsigned short* dec_eW1o = prep_c(dec_e.W1, 128*128);

    // GEMM wrappers ------------------------------------------------------
    auto mlp_hidden = [&](const float* A, int M, int K, const unsigned short* Wt,
                          const float* b, float* o) {
        k_gemm<true,false><<<dim3(M/128,1),256,0,stream>>>(
            A, K, Wt, K, b, nullptr, nullptr, o, 128, nullptr, nullptr, K);
    };
    auto mlp_lastLN = [&](const float* A, int M, const unsigned short* Wt, const float* b,
                          const float* res, float* o, float* sy, float* srs) {
        k_gemm<false,true><<<dim3(M/128,1),256,0,stream>>>(
            A, 128, Wt, 128, b, res, nullptr, o, 128, sy, srs, 128);
    };
    auto gemm_bT = [&](const float* A, int M, const unsigned short* Wo, int N,
                       const float* mask, float* o) {   // o = A @ W^T  (K=128)
        k_gemm<false,false><<<dim3(M/128,N/128),256,0,stream>>>(
            A, 128, Wo, 128, nullptr, nullptr, mask, o, N, nullptr, nullptr, 128);
    };

    float *e_fin = nullptr, *n_fin = nullptr;

    // forward GNN pass; stashes activations for input-grad backward -------
    auto forward = [&](const float* ef, bool head) {
        k_affine1<<<ew(NDIFFv*LATv),256,0,stream>>>(ef, enc_e.W0, enc_e.b0, h1_enc, NDIFFv);
        mlp_lastLN(h1_enc, NDIFFv, enc_eW1t, enc_e.b1, nullptr, eA, y_enc, rstd_enc);
        k_affine1<<<ew(NNODEv*LATv),256,0,stream>>>(nodes, enc_n.W0, enc_n.b0, tmp1, NNODEv);
        mlp_lastLN(tmp1, NNODEv, enc_nW1t, enc_n.b1, nullptr, nA, nullptr, nullptr);
        float *ec = eA, *en = eB, *nc = nA, *nn = nB;
        for (int i = 0; i < 5; ++i) {
            k_build_msg<<<ew(NDIFFv*LATv),256,0,stream>>>(ec, nc, senders, receivers, msg);
            mlp_hidden(msg, NDIFFv, 384, mp_eW0t, mp_e.b0, h1_e[i]);
            mlp_lastLN(h1_e[i], NDIFFv, mp_eW1t, mp_e.b1, ec, en, y_e[i], rstd_e[i]);
            { float* t = ec; ec = en; en = t; }
            (void)hipMemsetAsync(agg, 0, (size_t)NNODEv*LATv*4, stream);
            k_scatter_add<<<ew(NDIFFv*LATv),256,0,stream>>>(ec, 128, 0, receivers, agg, NDIFFv);
            k_build_cat<<<ew(NNODEv*LATv),256,0,stream>>>(nc, agg, cat);
            mlp_hidden(cat, NNODEv, 256, mp_nW0t, mp_n.b0, h1_n[i]);
            mlp_lastLN(h1_n[i], NNODEv, mp_nW1t, mp_n.b1, nc, nn, y_n[i], rstd_n[i]);
            { float* t = nc; nc = nn; nn = t; }
        }
        e_fin = ec; n_fin = nc;
        mlp_hidden(e_fin, NDIFFv, 128, dec_eW0t, dec_e.b0, h1_d);
        mlp_hidden(h1_d,  NDIFFv, 128, dec_eW1t, dec_e.b1, h2_d);
        k_rowdot<<<NDIFFv,128,0,stream>>>(h2_d, dec_e.W2, dec_e.b2, dec_ev);
        if (head) {
            k_sum<<<1,256,0,stream>>>(dec_ev, NDIFFv, out + NSTATEv);   // H
            mlp_hidden(n_fin, NNODEv, 128, dec_nW0t, dec_n.b0, tmp1);
            mlp_hidden(tmp1,  NNODEv, 128, dec_nW1t, dec_n.b1, tmp2);
            k_rowdot<<<NNODEv,128,0,stream>>>(tmp2, dec_n.W2, dec_n.b2, next_y);
        }
    };

    // backward: dH = d(sum dec_e)/d(ef)  -- input gradients only ----------
    auto backward = [&]() {
        k_seed_dec<<<ew(NDIFFv*LATv),256,0,stream>>>(h2_d, dec_e.W2, tmp1, NDIFFv);
        gemm_bT(tmp1, NDIFFv, dec_eW1o, 128, h1_d, tmp2);
        gemm_bT(tmp2, NDIFFv, dec_eW0o, 128, nullptr, g_e);
        (void)hipMemsetAsync(g_n, 0, (size_t)NNODEv*LATv*4, stream);
        for (int i = 4; i >= 0; --i) {
            k_ln_bwd<<<NNODEv,128,0,stream>>>(y_n[i], rstd_n[i], g_n, tmp1);
            gemm_bT(tmp1, NNODEv, mp_nW1o, 128, h1_n[i], tmp2);
            gemm_bT(tmp2, NNODEv, mp_nW0o, 256, nullptr, cat);          // g_cat
            k_add_slice<<<ew(NNODEv*LATv),256,0,stream>>>(g_n, cat, 256, 0, NNODEv);
            k_gather_add<<<ew(NDIFFv*LATv),256,0,stream>>>(g_e, cat, 256, 128,
                                                           receivers, NDIFFv);
            k_ln_bwd<<<NDIFFv,128,0,stream>>>(y_e[i], rstd_e[i], g_e, tmp1);
            gemm_bT(tmp1, NDIFFv, mp_eW1o, 128, h1_e[i], tmp2);
            gemm_bT(tmp2, NDIFFv, mp_eW0o, 384, nullptr, msg);          // g_msg
            k_add_slice<<<ew(NDIFFv*LATv),256,0,stream>>>(g_e, msg, 384, 0, NDIFFv);
            k_scatter_add<<<ew(NDIFFv*LATv),256,0,stream>>>(msg, 384, 128, senders, g_n, NDIFFv);
            k_scatter_add<<<ew(NDIFFv*LATv),256,0,stream>>>(msg, 384, 256, receivers, g_n, NDIFFv);
        }
        k_ln_bwd<<<NDIFFv,128,0,stream>>>(y_enc, rstd_enc, g_e, tmp1);
        gemm_bT(tmp1, NDIFFv, enc_eW1o, 128, h1_enc, tmp2);
        k_rowdot<<<NDIFFv,128,0,stream>>>(tmp2, enc_e.W0, nullptr, dH);
    };

    // f(upd) assembly + dyn step: xn = xc + DT * U_inv @ (L_inv @ (P_inv @ f))[:NDIFF]
    auto dyn_step = [&](const float* xc, float* xn) {
        k_matvec<<<NNODEv,256,0,stream>>>(AL, dH + NQv, F0v, -1.f, NNODEv, fvec); // F0
        k_copy<<<ew(NNODEv),256,0,stream>>>(fvec + 2048, F1v, NNODEv);            // F1
        k_sub<<<ew(NQv),256,0,stream>>>(fvec + 4096, dH, ACtv, NQv);              // F2
        k_matvec<<<NSTATEv,256,0,stream>>>(P_inv, fvec, nullptr, 1.f, NSTATEv, hvec);
        k_matvec<<<NSTATEv,256,0,stream>>>(L_inv, hvec, nullptr, 1.f, NSTATEv, wvec);
        k_matvec<<<NDIFFv,256,0,stream>>>(U_inv, wvec, xc, DTv, NDIFFv, xn);
    };

    // ------------------------------- pipeline -------------------------------
    forward(edges, true);                                   // H, next_y, stash@x0
    // e_volt == next_y in every f() evaluation: precompute once
    k_matvec_t<<<NRv/128,128,0,stream>>>(AR, next_y, 1.f, NNODEv, NRv, t_r);      // AR^T y
    k_matvec<<<NNODEv,256,0,stream>>>(AR, t_r, nullptr, -1.f, NRv, F0v);          // -AR AR^T y
    k_matvec_t<<<NNODEv/128,128,0,stream>>>(AL, next_y, 1.f, NNODEv, NNODEv, F1v);// AL^T y
    k_matvec_t<<<NQv/128,128,0,stream>>>(AC, next_y, 1.f, NNODEv, NQv, ACtv);     // AC^T y

    backward();                 // dH @ x0
    dyn_step(edges, x1);        // x1 = x0 + DT*dyn
    forward(x1, false);
    backward();                 // dH @ x1
    dyn_step(x1, x2);           // x2
    forward(x2, false);
    backward();                 // dH @ x2 (for residual)

    k_copy<<<ew(NDIFFv),256,0,stream>>>(out, x2, NDIFFv);
    k_copy<<<ew(NNODEv),256,0,stream>>>(out + NDIFFv, next_y, NNODEv);
    k_abs_sub<<<ew(NQv),256,0,stream>>>(out + NSTATEv + 1, dH, ACtv, NQv); // |F2|
}
